// DescrptDPA1_82514911691362
// MI455X (gfx1250) — matrix operand order repaired and checked by reading, not compiled, not run on hardware
//
#include <hip/hip_runtime.h>

typedef __attribute__((ext_vector_type(16))) _Float16 v16h;
typedef __attribute__((ext_vector_type(8)))  float    v8f;
typedef unsigned int u32;
typedef __attribute__((ext_vector_type(4))) unsigned int u32x4;

namespace {
constexpr int kNLOC   = 4096;
constexpr int kNNEI   = 120;
constexpr int kKP     = 128;   // padded neighbor count
constexpr int kM      = 100;   // feature dim
constexpr int kMP     = 128;   // padded feature dim
constexpr int kATTN   = 128;
constexpr int kNLAYER = 2;
constexpr int kTEBD   = 8;
constexpr int kAXIS   = 16;
constexpr int kLDH    = 136;   // f16 LDS row stride (halves) -> 272B = 17*16, keeps 16B align
constexpr int kLDC    = 132;   // f32 LDS row stride (floats)
constexpr float kRCUT = 6.0f, kRCUT_SMTH = 0.5f;
constexpr float kLN_EPS = 1e-5f;
constexpr int kOUT_PER_ATOM = kM * kAXIS + kTEBD;   // 1608
constexpr float kINV_TEMP = 0.08838834764831845f;   // 1/sqrt(128)

// embedding weight pack (floats), aliased into s_w before layer 0
constexpr int W0_OFF = 0;      // 17x25
constexpr int B0_OFF = 425;    // 25
constexpr int W1_OFF = 450;    // 25x50
constexpr int B1_OFF = 1700;   // 50
constexpr int W2_OFF = 1750;   // 50x100
constexpr int B2_OFF = 6750;   // 100

constexpr size_t kSmemBytes =
    size_t(5) * kKP * kLDH * sizeof(_Float16) +   // s_g, s_q, s_k, s_v, s_w
    size_t(kKP) * kLDC * sizeof(float) +          // sc (f32 scratch: pre-norm QKV, scores, O)
    size_t(kKP) * sizeof(float) * 2 +             // sr, sw*m
    size_t(kKP) * 4 * sizeof(float) +             // rhat (xyz + pad)
    size_t(kKP) * sizeof(u32) +                   // neighbor mask
    size_t(4) * kM * sizeof(float);               // gr
} // namespace

__device__ __forceinline__ u32 pack2h(float a, float b) {
  union { _Float16 h[2]; u32 u; } t;
  t.h[0] = (_Float16)a; t.h[1] = (_Float16)b;
  return t.u;
}

// ---- WMMA fragment helpers (CDNA5 16x16x32 f16 layouts, wave32) -------------
// All fragment base addresses are 16B aligned (row stride 272B, K offsets are
// multiples of 16B), so each fragment is exactly two ds_load_b128.

// A matrix 16x32 (MxK), row-major in LDS with stride lda (halves).
// lanes 0-15: M=lane, K base 0 ; lanes 16-31: M=lane-16, K base +8.
// VGPR j(0..3): K = base+2j,2j+1 ; VGPR j(4..7): K = base+16+2(j-4)+{0,1}.
__device__ __forceinline__ v16h load_a_frag(const _Float16* A, int lda,
                                            int row0, int col0, int lane) {
  const _Float16* rp = A + (row0 + (lane & 15)) * lda + col0 + ((lane & 16) ? 8 : 0);
  v16h a;
  u32x4* au = reinterpret_cast<u32x4*>(&a);
  au[0] = *reinterpret_cast<const u32x4*>(rp);        // K pairs 0..7   (16B)
  au[1] = *reinterpret_cast<const u32x4*>(rp + 16);   // K pairs 16..23 (16B)
  return a;
}

// B matrix 32x16 (KxN) taken from a transposed LDS image BT[n][k] (stride ldb).
// lanes 0-15: N=lane, K base 0 ; lanes 16-31: N=lane-16, K base +8 (same K order as A).
__device__ __forceinline__ v16h load_b_frag(const _Float16* BT, int ldb,
                                            int k0, int n0, int lane) {
  const _Float16* rp = BT + (n0 + (lane & 15)) * ldb + k0 + ((lane & 16) ? 8 : 0);
  v16h b;
  u32x4* bu = reinterpret_cast<u32x4*>(&b);
  bu[0] = *reinterpret_cast<const u32x4*>(rp);        // K base+0..7   (16B)
  bu[1] = *reinterpret_cast<const u32x4*>(rp + 16);   // K base+16..23 (16B)
  return b;
}

// One wave computes a 16x128 strip: rows [16*wrow, 16*wrow+16), 8 col tiles,
// inner dimension 128 walked in 4 k-steps of 32.
__device__ __forceinline__ void gemm_strip(const _Float16* A, int lda,
                                           const _Float16* BT, int ldb,
                                           int lane, int wrow, v8f acc[8]) {
#pragma unroll
  for (int ks = 0; ks < 4; ++ks) {
    v16h a = load_a_frag(A, lda, wrow * 16, ks * 32, lane);
#pragma unroll
    for (int j = 0; j < 8; ++j) {
      v16h b = load_b_frag(BT, ldb, ks * 32, j * 16, lane);
      acc[j] = __builtin_amdgcn_wmma_f32_16x16x32_f16(
          false, a, false, b, (short)0, acc[j], false, false);
    }
  }
}

// D/C layout: VGPR r -> M = r (lanes 0-15) or 8+r (lanes 16-31); N = lane&15.
__device__ __forceinline__ void store_strip_f32(float* D, int ldd, int lane,
                                                int wrow, const v8f acc[8], float scale) {
  int r0 = wrow * 16 + ((lane & 16) ? 8 : 0);
  int c0 = lane & 15;
#pragma unroll
  for (int j = 0; j < 8; ++j)
#pragma unroll
    for (int r = 0; r < 8; ++r)
      D[(r0 + r) * ldd + j * 16 + c0] = acc[j][r] * scale;
}

__device__ __forceinline__ void store_strip_f16(_Float16* D, int ldd, int lane,
                                                int wrow, const v8f acc[8]) {
  int r0 = wrow * 16 + ((lane & 16) ? 8 : 0);
  int c0 = lane & 15;
#pragma unroll
  for (int j = 0; j < 8; ++j)
#pragma unroll
    for (int r = 0; r < 8; ++r)
      D[(r0 + r) * ldd + j * 16 + c0] = (_Float16)acc[j][r];
}

// -----------------------------------------------------------------------------

extern "C" __global__ __launch_bounds__(256)
void dpa1_descriptor_kernel(const float* __restrict__ rij,
                            const unsigned char* __restrict__ nmask,
                            const int* __restrict__ atype,
                            const int* __restrict__ ntype,
                            const float* __restrict__ tebd,
                            const float* __restrict__ ew0, const float* __restrict__ eb0,
                            const float* __restrict__ ew1, const float* __restrict__ eb1,
                            const float* __restrict__ ew2, const float* __restrict__ eb2,
                            const float* __restrict__ wq, const float* __restrict__ bq,
                            const float* __restrict__ wk, const float* __restrict__ bk,
                            const float* __restrict__ wv, const float* __restrict__ bv,
                            const float* __restrict__ wo, const float* __restrict__ bo,
                            const float* __restrict__ lng, const float* __restrict__ lnb,
                            float* __restrict__ out) {
  extern __shared__ char smem[];
  _Float16* s_g = reinterpret_cast<_Float16*>(smem);          // [128][136] feature matrix
  _Float16* s_q = s_g + kKP * kLDH;                           // Q rows / AV(f16) rows
  _Float16* s_k = s_q + kKP * kLDH;                           // K rows / softmax probs rows
  _Float16* s_v = s_k + kKP * kLDH;                           // V TRANSPOSED [attn][nei]
  _Float16* s_w = s_v + kKP * kLDH;                           // staged weights (transposed)
  float* sc    = reinterpret_cast<float*>(s_w + kKP * kLDH);  // [128][132] f32 scratch
  float* s_sr  = sc + kKP * kLDC;                             // s(r) masked
  float* s_swm = s_sr + kKP;                                  // sw * mask
  float* s_rh  = s_swm + kKP;                                 // [128][4] rhat
  u32*   s_msk = reinterpret_cast<u32*>(s_rh + kKP * 4);      // neighbor mask
  float* s_gr  = reinterpret_cast<float*>(s_msk + kKP);       // [4][100]
  float* ew    = reinterpret_cast<float*>(s_w);               // alias: embedding weights

  const int atom = blockIdx.x;
  const int tid  = threadIdx.x;
  const int lane = tid & 31;
  const int wid  = tid >> 5;

  // Pull the (L2-resident) attention weights toward us early.
  if (lane == 0) {
    __builtin_prefetch(wq, 0, 2);
    __builtin_prefetch(wk, 0, 2);
    __builtin_prefetch(wv, 0, 2);
    __builtin_prefetch(wo, 0, 2);
  }

  // ---------------- Phase 0: embedding weights into LDS + neighbor geometry --
  for (int i = tid; i < 425;  i += 256) ew[W0_OFF + i] = ew0[i];
  for (int i = tid; i < 25;   i += 256) ew[B0_OFF + i] = eb0[i];
  for (int i = tid; i < 1250; i += 256) ew[W1_OFF + i] = ew1[i];
  for (int i = tid; i < 50;   i += 256) ew[B1_OFF + i] = eb1[i];
  for (int i = tid; i < 5000; i += 256) ew[W2_OFF + i] = ew2[i];
  for (int i = tid; i < 100;  i += 256) ew[B2_OFF + i] = eb2[i];

  if (tid < kKP) {
    const int k = tid;
    float srv = 0.f, swmv = 0.f, rhx = 0.f, rhy = 0.f, rhz = 0.f;
    u32 mk = 0;
    if (k < kNNEI) {
      const float* rp = rij + (size_t(atom) * kNNEI + k) * 3;
      const float x = rp[0], y = rp[1], z = rp[2];
      const float r = sqrtf(x * x + y * y + z * z);
      const float m = nmask[size_t(atom) * kNNEI + k] ? 1.f : 0.f;
      float uu = (r - kRCUT_SMTH) / (kRCUT - kRCUT_SMTH);
      uu = fminf(fmaxf(uu, 0.f), 1.f);
      const float sw = uu * uu * uu * (-6.f * uu * uu + 15.f * uu - 10.f) + 1.f;
      const float inv_r = 1.f / r;
      srv = sw * inv_r * m;
      swmv = sw * m;
      rhx = x * inv_r; rhy = y * inv_r; rhz = z * inv_r;
      mk = (m > 0.f) ? 1u : 0u;
    }
    s_sr[k] = srv; s_swm[k] = swmv; s_msk[k] = mk;
    s_rh[k * 4 + 0] = rhx; s_rh[k * 4 + 1] = rhy;
    s_rh[k * 4 + 2] = rhz; s_rh[k * 4 + 3] = 0.f;
  }
  __syncthreads();

  // ---------------- Phase 1: embedding MLP (one thread per neighbor row) -----
  if (tid < kKP) {
    const int k = tid;
    _Float16* grow = s_g + k * kLDH;
    if (k < kNNEI) {
      float x[17];
      x[0] = s_sr[k];
      const int tn = ntype[size_t(atom) * kNNEI + k];
      const int tc = atype[atom];
#pragma unroll
      for (int j = 0; j < kTEBD; ++j) x[1 + j] = tebd[tn * kTEBD + j];
#pragma unroll
      for (int j = 0; j < kTEBD; ++j) x[9 + j] = tebd[tc * kTEBD + j];

      float h1[25];
#pragma unroll
      for (int o = 0; o < 25; ++o) {
        float a = ew[B0_OFF + o];
#pragma unroll
        for (int i = 0; i < 17; ++i) a += x[i] * ew[W0_OFF + i * 25 + o];
        h1[o] = tanhf(a);
      }
      float h2[50];
#pragma unroll
      for (int o = 0; o < 50; ++o) {
        float a = ew[B1_OFF + o];
#pragma unroll
        for (int i = 0; i < 25; ++i) a += h1[i] * ew[W1_OFF + i * 50 + o];
        h2[o] = tanhf(a) + h1[o % 25];          // resnet: + concat(h1,h1)
      }
      const float mval = s_msk[k] ? 1.f : 0.f;
#pragma unroll
      for (int o = 0; o < 100; o += 2) {        // paired -> u32 LDS stores
        float a0 = ew[B2_OFF + o], a1 = ew[B2_OFF + o + 1];
#pragma unroll
        for (int i = 0; i < 50; ++i) {
          a0 += h2[i] * ew[W2_OFF + i * 100 + o];
          a1 += h2[i] * ew[W2_OFF + i * 100 + o + 1];
        }
        const float g0 = (tanhf(a0) + h2[o % 50]) * mval;
        const float g1 = (tanhf(a1) + h2[(o + 1) % 50]) * mval;
        *reinterpret_cast<u32*>(grow + o) = pack2h(g0, g1);
      }
#pragma unroll
      for (int o = kM; o < kMP; o += 2)
        *reinterpret_cast<u32*>(grow + o) = 0u;   // zero padding cols
    } else {
      for (int o = 0; o < kMP; o += 2)
        *reinterpret_cast<u32*>(grow + o) = 0u;   // zero padded rows
    }
  }
  __syncthreads();

  // ---------------- Phase 2: attention layers (all GEMMs via WMMA) -----------
  for (int l = 0; l < kNLAYER; ++l) {
    // --- Q, K, V projections: C = g[128x128] @ W[128x128] ---
    for (int p = 0; p < 3; ++p) {
      const float* wm = (p == 0) ? wq : ((p == 1) ? wk : wv);
      const float* bb = ((p == 0) ? bq : ((p == 1) ? bk : bv)) + size_t(l) * kATTN;
      // stage W^T into s_w: s_w[a][m] = W[m][a]  (m>=100 zero-padded, harmless)
      for (int idx = tid; idx < kMP * kATTN; idx += 256) {
        const int mI = idx & (kMP - 1);
        const int aI = idx >> 7;
        const float val = (mI < kM)
            ? wm[(size_t(l) * kM + mI) * kATTN + aI] : 0.f;
        s_w[aI * kLDH + mI] = (_Float16)val;
      }
      __syncthreads();
      {
        v8f acc[8];
        const v8f vz = {0.f, 0.f, 0.f, 0.f, 0.f, 0.f, 0.f, 0.f};
#pragma unroll
        for (int j = 0; j < 8; ++j) acc[j] = vz;
        gemm_strip(s_g, kLDH, s_w, kLDH, lane, wid, acc);
        store_strip_f32(sc, kLDC, lane, wid, acc, 1.0f);
      }
      __syncthreads();
      // row-wise bias + l2norm, write f16 (V is written transposed)
      if (tid < kKP) {
        const int row = tid;
        float ss = 0.f;
        for (int c = 0; c < kATTN; ++c) {
          const float t = sc[row * kLDC + c] + bb[c];
          ss += t * t;
        }
        const float inv = 1.f / fmaxf(sqrtf(ss), 1e-12f);
        if (p == 0) {
          for (int c = 0; c < kATTN; c += 2)
            *reinterpret_cast<u32*>(s_q + row * kLDH + c) =
                pack2h((sc[row * kLDC + c] + bb[c]) * inv,
                       (sc[row * kLDC + c + 1] + bb[c + 1]) * inv);
        } else if (p == 1) {
          for (int c = 0; c < kATTN; c += 2)
            *reinterpret_cast<u32*>(s_k + row * kLDH + c) =
                pack2h((sc[row * kLDC + c] + bb[c]) * inv,
                       (sc[row * kLDC + c + 1] + bb[c + 1]) * inv);
        } else {
          for (int c = 0; c < kATTN; ++c)      // transposed: scatter, scalar stores
            s_v[c * kLDH + row] = (_Float16)((sc[row * kLDC + c] + bb[c]) * inv);
        }
      }
      __syncthreads();
    }

    // --- scores = Qn @ Kn^T / temp : A = s_q rows, BT image = s_k rows ---
    {
      v8f acc[8];
      const v8f vz = {0.f, 0.f, 0.f, 0.f, 0.f, 0.f, 0.f, 0.f};
#pragma unroll
      for (int j = 0; j < 8; ++j) acc[j] = vz;
      gemm_strip(s_q, kLDH, s_k, kLDH, lane, wid, acc);
      store_strip_f32(sc, kLDC, lane, wid, acc, kINV_TEMP);
    }
    __syncthreads();

    // --- masked softmax + switch + angular gate; probs (f16) overwrite s_k ---
    if (tid < kKP) {
      const int qr = tid;
      const u32 mq = s_msk[qr];
      float mx = -3.4e38f;
      for (int kk = 0; kk < kKP; ++kk) {
        const bool valid = mq && s_msk[kk];
        const float s = valid ? sc[qr * kLDC + kk] : -3.0e38f;
        mx = fmaxf(mx, s);
      }
      float sum = 0.f;
      for (int kk = 0; kk < kKP; ++kk) {
        const bool valid = mq && s_msk[kk];
        const float e = valid ? expf(sc[qr * kLDC + kk] - mx) : 0.f;
        sum += e;
        sc[qr * kLDC + kk] = e;
      }
      const float inv = (sum > 0.f) ? 1.f / sum : 0.f;
      const float swq = s_swm[qr];
      const float rqx = s_rh[qr * 4], rqy = s_rh[qr * 4 + 1], rqz = s_rh[qr * 4 + 2];
      for (int kk = 0; kk < kKP; kk += 2) {
        const float g0 = rqx * s_rh[kk * 4] + rqy * s_rh[kk * 4 + 1]
                       + rqz * s_rh[kk * 4 + 2];
        const float g1 = rqx * s_rh[kk * 4 + 4] + rqy * s_rh[kk * 4 + 5]
                       + rqz * s_rh[kk * 4 + 6];
        const float p0 = sc[qr * kLDC + kk] * inv * swq * s_swm[kk] * g0;
        const float p1 = sc[qr * kLDC + kk + 1] * inv * swq * s_swm[kk + 1] * g1;
        *reinterpret_cast<u32*>(s_k + qr * kLDH + kk) = pack2h(p0, p1);
      }
    }
    // stage wo^T into s_w (free after V projection): s_w[m][a] = wo[a][m]
    for (int idx = tid; idx < kATTN * kMP; idx += 256) {
      const int aI = idx & (kATTN - 1);
      const int mI = idx >> 7;
      const float val = (mI < kM)
          ? wo[(size_t(l) * kATTN + aI) * kM + mI] : 0.f;
      s_w[mI * kLDH + aI] = (_Float16)val;
    }
    __syncthreads();

    // --- AV = P @ V : A = probs rows (s_k), BT image = s_v[a][nei]; f16 -> s_q
    {
      v8f acc[8];
      const v8f vz = {0.f, 0.f, 0.f, 0.f, 0.f, 0.f, 0.f, 0.f};
#pragma unroll
      for (int j = 0; j < 8; ++j) acc[j] = vz;
      gemm_strip(s_k, kLDH, s_v, kLDH, lane, wid, acc);
      store_strip_f16(s_q, kLDH, lane, wid, acc);
    }
    __syncthreads();

    // --- O = AV @ wo : A = s_q rows, BT image = s_w[m][a]; f32 -> sc ---
    {
      v8f acc[8];
      const v8f vz = {0.f, 0.f, 0.f, 0.f, 0.f, 0.f, 0.f, 0.f};
#pragma unroll
      for (int j = 0; j < 8; ++j) acc[j] = vz;
      gemm_strip(s_q, kLDH, s_w, kLDH, lane, wid, acc);
      store_strip_f32(sc, kLDC, lane, wid, acc, 1.0f);
    }
    __syncthreads();

    // --- bias + LayerNorm(M=100) + residual into g (cols < 100 only) ---
    if (tid < kKP) {
      const int row = tid;
      const float* bb = bo + size_t(l) * kM;
      const float* gg = lng + size_t(l) * kM;
      const float* bbeta = lnb + size_t(l) * kM;
      float mu = 0.f;
      for (int c = 0; c < kM; ++c) mu += sc[row * kLDC + c] + bb[c];
      mu *= (1.f / kM);
      float var = 0.f;
      for (int c = 0; c < kM; ++c) {
        const float d = sc[row * kLDC + c] + bb[c] - mu;
        var += d * d;
      }
      var *= (1.f / kM);
      const float inv = rsqrtf(var + kLN_EPS);
      for (int c = 0; c < kM; c += 2) {
        const float t0 = sc[row * kLDC + c] + bb[c];
        const float t1 = sc[row * kLDC + c + 1] + bb[c + 1];
        const float o0 = (t0 - mu) * inv * gg[c] + bbeta[c];
        const float o1 = (t1 - mu) * inv * gg[c + 1] + bbeta[c + 1];
        u32* gp = reinterpret_cast<u32*>(s_g + row * kLDH + c);
        union { u32 u; _Float16 h[2]; } old;
        old.u = *gp;
        *gp = pack2h((float)old.h[0] + o0, (float)old.h[1] + o1);
      }
    }
    __syncthreads();
  }

  // ---------------- Phase 3: g *= m ; gr = env^T g / NNEI ; d = gr gr[:,:16]^T
  if (tid < kKP) {
    if (!s_msk[tid]) {
      for (int c = 0; c < kM; c += 2)
        *reinterpret_cast<u32*>(s_g + tid * kLDH + c) = 0u;
    }
  }
  __syncthreads();

  for (int idx = tid; idx < 4 * kM; idx += 256) {
    const int c = idx / kM;
    const int mm = idx % kM;
    float acc = 0.f;
    for (int k = 0; k < kNNEI; ++k) {
      const float envv = (c == 0) ? s_sr[k] : s_sr[k] * s_rh[k * 4 + (c - 1)];
      acc += envv * (float)s_g[k * kLDH + mm];
    }
    s_gr[c * kM + mm] = acc * (1.0f / kNNEI);
  }
  __syncthreads();

  float* op = out + size_t(atom) * kOUT_PER_ATOM;
  for (int idx = tid; idx < kM * kAXIS; idx += 256) {
    const int mm = idx >> 4;
    const int a  = idx & (kAXIS - 1);
    float d = 0.f;
#pragma unroll
    for (int c = 0; c < 4; ++c) d += s_gr[c * kM + mm] * s_gr[c * kM + a];
    op[idx] = d;
  }
  if (tid < kTEBD) op[kM * kAXIS + tid] = tebd[atype[atom] * kTEBD + tid];
}

// -----------------------------------------------------------------------------

extern "C" void kernel_launch(void* const* d_in, const int* in_sizes, int n_in,
                              void* d_out, int out_size, void* d_ws, size_t ws_size,
                              hipStream_t stream) {
  (void)in_sizes; (void)n_in; (void)out_size; (void)d_ws; (void)ws_size;
  const float* rij   = (const float*)d_in[0];
  const unsigned char* nmask = (const unsigned char*)d_in[1];   // jnp bool -> 1 byte
  const int* atype   = (const int*)d_in[2];
  const int* ntype   = (const int*)d_in[3];
  const float* tebd  = (const float*)d_in[4];
  const float* ew0   = (const float*)d_in[5];
  const float* eb0   = (const float*)d_in[6];
  const float* ew1   = (const float*)d_in[7];
  const float* eb1   = (const float*)d_in[8];
  const float* ew2   = (const float*)d_in[9];
  const float* eb2   = (const float*)d_in[10];
  const float* wq    = (const float*)d_in[11];
  const float* bq    = (const float*)d_in[12];
  const float* wk    = (const float*)d_in[13];
  const float* bk    = (const float*)d_in[14];
  const float* wv    = (const float*)d_in[15];
  const float* bv    = (const float*)d_in[16];
  const float* wo    = (const float*)d_in[17];
  const float* bo    = (const float*)d_in[18];
  const float* lng   = (const float*)d_in[19];
  const float* lnb   = (const float*)d_in[20];
  float* outp = (float*)d_out;

  dpa1_descriptor_kernel<<<kNLOC, 256, kSmemBytes, stream>>>(
      rij, nmask, atype, ntype, tebd,
      ew0, eb0, ew1, eb1, ew2, eb2,
      wq, bq, wk, bk, wv, bv, wo, bo, lng, lnb, outp);
}
